// SelectiveSSM_69355131895937
// MI455X (gfx1250) — matrix operand order repaired and checked by reading, not compiled, not run on hardware
//
#include <hip/hip_runtime.h>

// ---------------------------------------------------------------------------
// Selective SSM for MI455X (gfx1250, wave32, WMMA, async-LDS double buffering).
//   1. pack_bf16        : x fp32 -> bf16
//   2. pack_w_transpose : W (N,K) fp32 -> Wt (K,N) bf16   (x3 weights)
//   3. pack_bm_transpose: Bm (16,K) -> Bmt (K,16) bf16
//   4. gemm_gate        : u = (x@W2^T + b2) * sigmoid(x@Wg^T + bg)  [WMMA]
//   5. uB_kernel        : uB = u @ Bm^T                             [WMMA]
//   6. apow/scan_local/scan_carry : chunked parallel scan (depth 4096 -> ~128)
//   7. y_fix_proj       : h = local + A^{off+1}@carry ; y = h @ Cm^T
//   8. gemm_out         : out = y @ W_out^T + b_out                 [WMMA]
// ---------------------------------------------------------------------------

#define DIMD    1024
#define NSTATE  16
#define BB      4
#define SS      4096
#define MTOK    (BB * SS)
#define LCHUNK  64
#define CCHUNK  (SS / LCHUNK)
#define NCH     (DIMD / 32)        // K chunks of 32

#if defined(__has_builtin)
#if __has_builtin(__builtin_amdgcn_global_load_async_to_lds_b128) && \
    __has_builtin(__builtin_amdgcn_s_wait_asynccnt)
#define USE_ASYNC 1
#endif
#endif
#ifndef USE_ASYNC
#define USE_ASYNC 0
#endif

typedef __attribute__((ext_vector_type(4)))  int    v4i;
typedef __attribute__((ext_vector_type(16))) __bf16 v16bf;
typedef __attribute__((ext_vector_type(8)))  float  v8f;

#if USE_ASYNC
#define NBUF 2
// builtin wants v4i* in AS1 (global) / AS3 (LDS)
#define GPTR(p) ((__attribute__((address_space(1))) v4i*)(void*)(p))
#define LPTR(p) ((__attribute__((address_space(3))) v4i*)(void*)(p))
#else
#define NBUF 1
#endif

union FragU { uint4 u[2]; v16bf v; };

__device__ __forceinline__ unsigned short f2bf(float f) {
    unsigned int u = __float_as_uint(f);
    u += 0x7FFFu + ((u >> 16) & 1u);
    return (unsigned short)(u >> 16);
}

// A-frag: lane (m = l%16, half = l/16): 8 bf16 at K=half*8, 8 at K=16+half*8
__device__ __forceinline__ v16bf load_frag_a(const unsigned short* rowBase, int half) {
    FragU f;
    f.u[0] = *(const uint4*)(rowBase + half * 8);
    f.u[1] = *(const uint4*)(rowBase + 16 + half * 8);
    return f.v;
}
// B-frag: lane (n = l%16, half = l/16): 8 bf16 at K=half*8, 8 at K=16+half*8 (strided column)
__device__ __forceinline__ v16bf load_frag_b(const unsigned short* colBase, int stride, int half) {
    union { unsigned short e[16]; v16bf v; } f;
#pragma unroll
    for (int s = 0; s < 16; ++s)
        f.e[s] = colBase[((s < 8) ? half * 8 + s : 16 + half * 8 + (s - 8)) * stride];
    return f.v;
}
__device__ __forceinline__ v8f wmma_bf16(v16bf a, v16bf b, v8f c) {
    return __builtin_amdgcn_wmma_f32_16x16x32_bf16(false, a, false, b, (short)0, c,
                                                   false, false);
}

// --------------------------- packing ---------------------------------------
__global__ __launch_bounds__(256) void pack_bf16(const float* __restrict__ src,
                                                 unsigned short* __restrict__ dst,
                                                 int n) {
    int i = blockIdx.x * blockDim.x + threadIdx.x;
    int stride = gridDim.x * blockDim.x;
    for (; i < n; i += stride) dst[i] = f2bf(src[i]);
}

// (N,K) fp32 row-major -> (K,N) bf16 row-major, 32x32 LDS tiles
__global__ __launch_bounds__(256) void pack_w_transpose(
    const float* __restrict__ src, unsigned short* __restrict__ dst) {
    __shared__ float tile[32][33];
    const int x = threadIdx.x & 31, y8 = threadIdx.x >> 5;
    const int kBase = blockIdx.x * 32, nBase = blockIdx.y * 32;
#pragma unroll
    for (int i = 0; i < 4; ++i) {
        int y = y8 * 4 + i;
        tile[y][x] = src[(size_t)(nBase + y) * DIMD + kBase + x];
    }
    __syncthreads();
#pragma unroll
    for (int i = 0; i < 4; ++i) {
        int y = y8 * 4 + i;
        dst[(size_t)(kBase + y) * DIMD + nBase + x] = f2bf(tile[x][y]);
    }
}

// Bm (16,K) fp32 -> Bmt (K,16) bf16
__global__ __launch_bounds__(256) void pack_bm_transpose(
    const float* __restrict__ src, unsigned short* __restrict__ dst) {
    int idx = blockIdx.x * 256 + threadIdx.x;      // K*16 = 16384
    int k = idx >> 4, n = idx & 15;
    dst[idx] = f2bf(src[(size_t)n * DIMD + k]);
}

// --------------------------- GEMM tiles -------------------------------------
// Block 256 thr (8 waves). Tile 256(M) x 64(N), K-chunks of 32.
// Wave owns 32 M-rows (2 A-frags) x full 64 N.
#define XS_STRIDE 40   // bf16 (80 B rows, 16B aligned frag reads)
#define WS_STRIDE 72   // bf16 (144 B rows)

// stage one K-chunk: X tile 256x32 (4 b128/thread), Wt tile 32x64 (1 b128/thread/matrix)
#if USE_ASYNC
__device__ __forceinline__ void stage_x_async(const unsigned short* xg,
                                              unsigned short* xsb,
                                              int tid, int mBase, int k0) {
    const unsigned short* g = xg + (size_t)(mBase + tid) * DIMD + k0;
    unsigned short* l = xsb + tid * XS_STRIDE;
#pragma unroll
    for (int i = 0; i < 4; ++i)
        __builtin_amdgcn_global_load_async_to_lds_b128(GPTR(g + i * 8),
                                                       LPTR(l + i * 8), 0, 0);
}
__device__ __forceinline__ void stage_w_async(const unsigned short* wt,
                                              unsigned short* wsb,
                                              int tid, int nBase, int k0) {
    int k = tid >> 3, nsub = (tid & 7) * 8;
    __builtin_amdgcn_global_load_async_to_lds_b128(
        GPTR(wt + (size_t)(k0 + k) * DIMD + nBase + nsub),
        LPTR(wsb + k * WS_STRIDE + nsub), 0, 0);
}
#else
__device__ __forceinline__ void stage_x_sync(const unsigned short* xg,
                                             unsigned short* xsb,
                                             int tid, int mBase, int k0) {
    const uint4* g = (const uint4*)(xg + (size_t)(mBase + tid) * DIMD + k0);
    uint4 v0 = g[0], v1 = g[1], v2 = g[2], v3 = g[3];
    *(uint4*)(xsb + tid * XS_STRIDE) = v0;
    *(uint4*)(xsb + tid * XS_STRIDE + 8) = v1;
    *(uint4*)(xsb + tid * XS_STRIDE + 16) = v2;
    *(uint4*)(xsb + tid * XS_STRIDE + 24) = v3;
    if (k0 + 32 < DIMD)
        __builtin_prefetch(xg + (size_t)(mBase + tid) * DIMD + k0 + 32, 0, 1);
}
__device__ __forceinline__ void stage_w_sync(const unsigned short* wt,
                                             unsigned short* wsb,
                                             int tid, int nBase, int k0) {
    int k = tid >> 3, nsub = (tid & 7) * 8;
    uint4 v = *(const uint4*)(wt + (size_t)(k0 + k) * DIMD + nBase + nsub);
    *(uint4*)(wsb + k * WS_STRIDE + nsub) = v;
}
#endif

// --------------------------- 4. gated dual GEMM -----------------------------
__device__ __forceinline__ void gate_compute(const unsigned short* xsb,
                                             const unsigned short* w2sb,
                                             const unsigned short* wgsb,
                                             int wave, int lane,
                                             v8f (&acc2)[2][4], v8f (&accg)[2][4]) {
    const int m = lane & 15, half = lane >> 4;
    v16bf a0 = load_frag_a(xsb + (wave * 32 + m) * XS_STRIDE, half);
    v16bf a1 = load_frag_a(xsb + (wave * 32 + 16 + m) * XS_STRIDE, half);
    v16bf bf[4];
#pragma unroll
    for (int s = 0; s < 4; ++s) bf[s] = load_frag_b(w2sb + m + s * 16, WS_STRIDE, half);
#pragma unroll
    for (int s = 0; s < 4; ++s) {
        acc2[0][s] = wmma_bf16(a0, bf[s], acc2[0][s]);
        acc2[1][s] = wmma_bf16(a1, bf[s], acc2[1][s]);
    }
#pragma unroll
    for (int s = 0; s < 4; ++s) bf[s] = load_frag_b(wgsb + m + s * 16, WS_STRIDE, half);
#pragma unroll
    for (int s = 0; s < 4; ++s) {
        accg[0][s] = wmma_bf16(a0, bf[s], accg[0][s]);
        accg[1][s] = wmma_bf16(a1, bf[s], accg[1][s]);
    }
}

__global__ __launch_bounds__(256) void gemm_gate(
    const unsigned short* __restrict__ xb,
    const unsigned short* __restrict__ w2t,
    const unsigned short* __restrict__ wgt,
    const float* __restrict__ b_in, const float* __restrict__ b_gate,
    unsigned short* __restrict__ ub) {
    __shared__ unsigned short xs[NBUF][256 * XS_STRIDE];
    __shared__ unsigned short w2s[NBUF][32 * WS_STRIDE];
    __shared__ unsigned short wgs[NBUF][32 * WS_STRIDE];
    const int tid = threadIdx.x, lane = tid & 31, wave = tid >> 5;
    const int mBase = blockIdx.x * 256, nBase = blockIdx.y * 64;
    v8f acc2[2][4] = {}, accg[2][4] = {};

#if USE_ASYNC
    stage_x_async(xb, xs[0], tid, mBase, 0);
    stage_w_async(w2t, w2s[0], tid, nBase, 0);
    stage_w_async(wgt, wgs[0], tid, nBase, 0);
    for (int it = 0; it < NCH - 1; ++it) {
        const int cur = it & 1;
        __syncthreads();                         // buf(1-cur) readers done
        stage_x_async(xb, xs[1 - cur], tid, mBase, (it + 1) * 32);
        stage_w_async(w2t, w2s[1 - cur], tid, nBase, (it + 1) * 32);
        stage_w_async(wgt, wgs[1 - cur], tid, nBase, (it + 1) * 32);
        __builtin_amdgcn_s_wait_asynccnt(6);     // cur chunk landed (in-order)
        __syncthreads();
        gate_compute(xs[cur], w2s[cur], wgs[cur], wave, lane, acc2, accg);
    }
    __syncthreads();
    __builtin_amdgcn_s_wait_asynccnt(0);
    __syncthreads();
    gate_compute(xs[(NCH - 1) & 1], w2s[(NCH - 1) & 1], wgs[(NCH - 1) & 1],
                 wave, lane, acc2, accg);
#else
    for (int it = 0; it < NCH; ++it) {
        __syncthreads();
        stage_x_sync(xb, xs[0], tid, mBase, it * 32);
        stage_w_sync(w2t, w2s[0], tid, nBase, it * 32);
        stage_w_sync(wgt, wgs[0], tid, nBase, it * 32);
        __syncthreads();
        gate_compute(xs[0], w2s[0], wgs[0], wave, lane, acc2, accg);
    }
#endif
    // epilogue: u = (p2 + b2) * sigmoid(pg + bg) -> bf16
    const int rowhi = (lane >> 4) * 8, ncol = lane & 15;
#pragma unroll
    for (int ms = 0; ms < 2; ++ms) {
#pragma unroll
        for (int s = 0; s < 4; ++s) {
            int n = nBase + s * 16 + ncol;
            float bi = b_in[DIMD + n], bg = b_gate[n];
#pragma unroll
            for (int r = 0; r < 8; ++r) {
                int t = mBase + wave * 32 + ms * 16 + rowhi + r;
                float p2 = acc2[ms][s][r] + bi;
                float pg = accg[ms][s][r] + bg;
                float gate = 1.0f / (1.0f + __expf(-pg));
                ub[(size_t)t * DIMD + n] = f2bf(p2 * gate);
            }
        }
    }
}

// --------------------------- 5. uB = u @ Bm^T (N=16) ------------------------
#define BS_STRIDE 24
__global__ __launch_bounds__(256) void uB_kernel(
    const unsigned short* __restrict__ ub,
    const unsigned short* __restrict__ bmt,     // (K,16) bf16
    float* __restrict__ uBp) {
    __shared__ unsigned short us[128 * XS_STRIDE];
    __shared__ unsigned short bs[32 * BS_STRIDE];
    const int tid = threadIdx.x, lane = tid & 31, wave = tid >> 5;
    const int mBase = blockIdx.x * 128;
    v8f acc = {};
    for (int k0 = 0; k0 < DIMD; k0 += 32) {
        __syncthreads();
        {
            int row = tid >> 1, kk = (tid & 1) * 16;
            const uint4* g = (const uint4*)(ub + (size_t)(mBase + row) * DIMD + k0 + kk);
            uint4 v0 = g[0], v1 = g[1];
            *(uint4*)(us + row * XS_STRIDE + kk) = v0;
            *(uint4*)(us + row * XS_STRIDE + kk + 8) = v1;
        }
        if (tid < 64) {
            int k = tid >> 1, nsub = (tid & 1) * 8;
            uint4 v = *(const uint4*)(bmt + (size_t)(k0 + k) * NSTATE + nsub);
            *(uint4*)(bs + k * BS_STRIDE + nsub) = v;
        }
        __syncthreads();
        v16bf a = load_frag_a(us + (wave * 16 + (lane & 15)) * XS_STRIDE, lane >> 4);
        v16bf b = load_frag_b(bs + (lane & 15), BS_STRIDE, lane >> 4);
        acc = wmma_bf16(a, b, acc);
    }
    const int rowhi = (lane >> 4) * 8, ncol = lane & 15;
#pragma unroll
    for (int r = 0; r < 8; ++r) {
        int t = mBase + wave * 16 + rowhi + r;
        uBp[(size_t)t * NSTATE + ncol] = acc[r];
    }
}

// --------------------------- 6. scan ----------------------------------------
__global__ __launch_bounds__(256) void apow_kernel(const float* __restrict__ A,
                                                   float* __restrict__ apow) {
    __shared__ float As[256], Pv[256];
    int tid = threadIdx.x, i = tid >> 4, j = tid & 15;
    As[tid] = A[tid];
    apow[tid] = (i == j) ? 1.0f : 0.0f;
    apow[256 + tid] = A[tid];
    Pv[tid] = A[tid];
    __syncthreads();
    for (int k = 2; k <= LCHUNK; ++k) {
        float s = 0.0f;
#pragma unroll
        for (int p = 0; p < 16; ++p) s += As[i * 16 + p] * Pv[p * 16 + j];
        __syncthreads();
        Pv[tid] = s;
        apow[(size_t)k * 256 + tid] = s;
        __syncthreads();
    }
}

__global__ __launch_bounds__(32) void scan_local(const float* __restrict__ A,
                                                 const float* __restrict__ uBp,
                                                 float* __restrict__ hsl,
                                                 float* __restrict__ finals) {
    const int b = blockIdx.x / CCHUNK, c = blockIdx.x % CCHUNK;
    const int i = threadIdx.x;
    const bool act = i < 16;
    float a[16];
#pragma unroll
    for (int j = 0; j < 16; ++j) a[j] = act ? A[i * 16 + j] : 0.0f;
    float h = 0.0f;
    const size_t base = ((size_t)b * SS + (size_t)c * LCHUNK) * NSTATE;
    for (int t = 0; t < LCHUNK; ++t) {
        float hn = act ? uBp[base + t * NSTATE + i] : 0.0f;
#pragma unroll
        for (int j = 0; j < 16; ++j) hn += a[j] * __shfl(h, j, 32);
        h = hn;
        if (act) hsl[base + t * NSTATE + i] = h;
    }
    if (act) finals[((size_t)b * CCHUNK + c) * NSTATE + i] = h;
}

__global__ __launch_bounds__(32) void scan_carry(const float* __restrict__ apow,
                                                 const float* __restrict__ finals,
                                                 float* __restrict__ carries) {
    const int b = blockIdx.x, i = threadIdx.x;
    const bool act = i < 16;
    const float* AL = apow + (size_t)LCHUNK * 256;
    float a[16];
#pragma unroll
    for (int j = 0; j < 16; ++j) a[j] = act ? AL[i * 16 + j] : 0.0f;
    float carry = 0.0f;
    for (int c = 0; c < CCHUNK; ++c) {
        if (act) carries[((size_t)b * CCHUNK + c) * NSTATE + i] = carry;
        float hn = act ? finals[((size_t)b * CCHUNK + c) * NSTATE + i] : 0.0f;
#pragma unroll
        for (int j = 0; j < 16; ++j) hn += a[j] * __shfl(carry, j, 32);
        carry = hn;
    }
}

// --------------------------- 7. fix-up + y = h @ Cm^T -----------------------
__global__ __launch_bounds__(256) void y_fix_proj(const float* __restrict__ apow,
                                                  const float* __restrict__ carries,
                                                  const float* __restrict__ hsl,
                                                  const float* __restrict__ Cm,
                                                  unsigned short* __restrict__ yb) {
    const int t = blockIdx.x;
    const int b = t / SS, si = t % SS, c = si / LCHUNK, toff = si % LCHUNK;
    __shared__ float hst[16];
    const int tid = threadIdx.x;
    if (tid < 16) {
        const float* Ap = apow + (size_t)(toff + 1) * 256 + tid * 16;
        const float* cr = carries + ((size_t)b * CCHUNK + c) * NSTATE;
        float s = hsl[(size_t)t * NSTATE + tid];
#pragma unroll
        for (int j = 0; j < 16; ++j) s += Ap[j] * cr[j];
        hst[tid] = s;
    }
    __syncthreads();
#pragma unroll
    for (int rep = 0; rep < 4; ++rep) {
        int d = tid + rep * 256;
        const float* cm = Cm + (size_t)d * NSTATE;
        float s = 0.0f;
#pragma unroll
        for (int n = 0; n < 16; ++n) s += hst[n] * cm[n];
        yb[(size_t)t * DIMD + d] = f2bf(s);
    }
}

// --------------------------- 8. out GEMM ------------------------------------
__device__ __forceinline__ void out_compute(const unsigned short* xsb,
                                            const unsigned short* wsb,
                                            int wave, int lane, v8f (&acc)[2][4]) {
    const int m = lane & 15, half = lane >> 4;
    v16bf a0 = load_frag_a(xsb + (wave * 32 + m) * XS_STRIDE, half);
    v16bf a1 = load_frag_a(xsb + (wave * 32 + 16 + m) * XS_STRIDE, half);
    v16bf bf[4];
#pragma unroll
    for (int s = 0; s < 4; ++s) bf[s] = load_frag_b(wsb + m + s * 16, WS_STRIDE, half);
#pragma unroll
    for (int s = 0; s < 4; ++s) {
        acc[0][s] = wmma_bf16(a0, bf[s], acc[0][s]);
        acc[1][s] = wmma_bf16(a1, bf[s], acc[1][s]);
    }
}

__global__ __launch_bounds__(256) void gemm_out(
    const unsigned short* __restrict__ yb,
    const unsigned short* __restrict__ wot,
    const float* __restrict__ b_out, float* __restrict__ out) {
    __shared__ unsigned short xs[NBUF][256 * XS_STRIDE];
    __shared__ unsigned short ws[NBUF][32 * WS_STRIDE];
    const int tid = threadIdx.x, lane = tid & 31, wave = tid >> 5;
    const int mBase = blockIdx.x * 256, nBase = blockIdx.y * 64;
    v8f acc[2][4] = {};

#if USE_ASYNC
    stage_x_async(yb, xs[0], tid, mBase, 0);
    stage_w_async(wot, ws[0], tid, nBase, 0);
    for (int it = 0; it < NCH - 1; ++it) {
        const int cur = it & 1;
        __syncthreads();
        stage_x_async(yb, xs[1 - cur], tid, mBase, (it + 1) * 32);
        stage_w_async(wot, ws[1 - cur], tid, nBase, (it + 1) * 32);
        __builtin_amdgcn_s_wait_asynccnt(5);
        __syncthreads();
        out_compute(xs[cur], ws[cur], wave, lane, acc);
    }
    __syncthreads();
    __builtin_amdgcn_s_wait_asynccnt(0);
    __syncthreads();
    out_compute(xs[(NCH - 1) & 1], ws[(NCH - 1) & 1], wave, lane, acc);
#else
    for (int it = 0; it < NCH; ++it) {
        __syncthreads();
        stage_x_sync(yb, xs[0], tid, mBase, it * 32);
        stage_w_sync(wot, ws[0], tid, nBase, it * 32);
        __syncthreads();
        out_compute(xs[0], ws[0], wave, lane, acc);
    }
#endif
    const int rowhi = (lane >> 4) * 8, ncol = lane & 15;
#pragma unroll
    for (int ms = 0; ms < 2; ++ms) {
#pragma unroll
        for (int s = 0; s < 4; ++s) {
            int n = nBase + s * 16 + ncol;
            float bo = b_out[n];
#pragma unroll
            for (int r = 0; r < 8; ++r) {
                int t = mBase + wave * 32 + ms * 16 + rowhi + r;
                out[(size_t)t * DIMD + n] = acc[ms][s][r] + bo;
            }
        }
    }
}

// ---------------------------------------------------------------------------
extern "C" void kernel_launch(void* const* d_in, const int* in_sizes, int n_in,
                              void* d_out, int out_size, void* d_ws, size_t ws_size,
                              hipStream_t stream) {
    const float* x      = (const float*)d_in[0];
    const float* W_in   = (const float*)d_in[1];
    const float* b_in   = (const float*)d_in[2];
    const float* A      = (const float*)d_in[3];
    const float* Bm     = (const float*)d_in[4];
    const float* Cm     = (const float*)d_in[5];
    const float* W_gate = (const float*)d_in[6];
    const float* b_gate = (const float*)d_in[7];
    const float* W_out  = (const float*)d_in[8];
    const float* b_out  = (const float*)d_in[9];
    float* out = (float*)d_out;

    char* wsp = (char*)d_ws;
    auto alloc = [&](size_t bytes) -> char* {
        char* p = wsp;
        wsp += (bytes + 255) & ~(size_t)255;
        return p;
    };
    unsigned short* xb   = (unsigned short*)alloc((size_t)MTOK * DIMD * 2);
    unsigned short* ubuf = (unsigned short*)alloc((size_t)MTOK * DIMD * 2);
    unsigned short* ybuf = (unsigned short*)alloc((size_t)MTOK * DIMD * 2);
    unsigned short* w2t  = (unsigned short*)alloc((size_t)DIMD * DIMD * 2);
    unsigned short* wgt  = (unsigned short*)alloc((size_t)DIMD * DIMD * 2);
    unsigned short* wot  = (unsigned short*)alloc((size_t)DIMD * DIMD * 2);
    unsigned short* bmt  = (unsigned short*)alloc((size_t)DIMD * NSTATE * 2);
    float* uBp     = (float*)alloc((size_t)MTOK * NSTATE * 4);
    float* hsl     = (float*)alloc((size_t)MTOK * NSTATE * 4);
    float* finals  = (float*)alloc((size_t)BB * CCHUNK * NSTATE * 4);
    float* carries = (float*)alloc((size_t)BB * CCHUNK * NSTATE * 4);
    float* apow    = (float*)alloc((size_t)(LCHUNK + 1) * 256 * 4);

    pack_bf16<<<4096, 256, 0, stream>>>(x, xb, MTOK * DIMD);
    pack_w_transpose<<<dim3(32, 32), 256, 0, stream>>>(W_in + (size_t)DIMD * DIMD, w2t);
    pack_w_transpose<<<dim3(32, 32), 256, 0, stream>>>(W_gate, wgt);
    pack_w_transpose<<<dim3(32, 32), 256, 0, stream>>>(W_out, wot);
    pack_bm_transpose<<<DIMD * NSTATE / 256, 256, 0, stream>>>(Bm, bmt);

    gemm_gate<<<dim3(MTOK / 256, DIMD / 64), 256, 0, stream>>>(xb, w2t, wgt, b_in,
                                                               b_gate, ubuf);
    uB_kernel<<<MTOK / 128, 256, 0, stream>>>(ubuf, bmt, uBp);
    apow_kernel<<<1, 256, 0, stream>>>(A, apow);
    scan_local<<<BB * CCHUNK, 32, 0, stream>>>(A, uBp, hsl, finals);
    scan_carry<<<BB, 32, 0, stream>>>(apow, finals, carries);
    y_fix_proj<<<MTOK, 256, 0, stream>>>(apow, carries, hsl, Cm, ybuf);
    gemm_out<<<dim3(MTOK / 256, DIMD / 64), 256, 0, stream>>>(ybuf, wot, b_out, out);
}
